// MoeLoopBlock_61959198212643
// MI455X (gfx1250) — compile-verified, matrix operand order checked by reading, not run on hardware
//
#include <hip/hip_runtime.h>
#include <math.h>

// Problem constants (match reference)
#define Bq 4
#define Sq 2048
#define Dq 1024
#define Fq 4096
#define Eq 8
#define Kq 2
#define Tq (Bq * Sq)       // 8192 tokens
#define NSLOT (Tq * Kq)    // 16384 (token, expert-slot) pairs

typedef __attribute__((ext_vector_type(16))) __bf16 v16bf;
typedef __attribute__((ext_vector_type(8)))  __bf16 v8bf;
typedef __attribute__((ext_vector_type(4)))  __bf16 v4bf;
typedef __attribute__((ext_vector_type(8)))  float  v8f;
typedef __attribute__((ext_vector_type(4)))  float  v4f;
typedef __attribute__((ext_vector_type(4)))  int    v4i;

// GEMM tiling: 128 threads = 4 wave32, wave grid 2(M) x 2(N)
#define BLK_M 128
#define BLK_N 64
#define BLK_K 32
#define KSTR  (BLK_K + 8)   // LDS row stride in halfs; 80B keeps 16B alignment, skews banks

// ---- CDNA5 async global->LDS path (guarded; falls back to VGPR staging) ----
#if defined(__has_builtin)
#if __has_builtin(__builtin_amdgcn_global_load_async_to_lds_b128)
#define HAVE_ASYNC_LDS 1
#endif
#endif

#ifdef HAVE_ASYNC_LDS
// Builtin expects int4 pointers: global (addrspace 1) source, LDS (addrspace 3) dest.
#define GPTR(p) ((__attribute__((address_space(1))) v4i*)((__bf16*)(p)))
#define LPTR(p) ((__attribute__((address_space(3))) v4i*)((__bf16*)(p)))
__device__ __forceinline__ void wait_async0() {
#if __has_builtin(__builtin_amdgcn_s_wait_asynccnt)
    __builtin_amdgcn_s_wait_asynccnt(0);
#else
    asm volatile("s_wait_asynccnt 0x0" ::: "memory");
#endif
}
#endif

// ---------------------------------------------------------------------------
// Load one 16x32 bf16 WMMA operand fragment from an LDS tile stored row-major
// ([row][K], stride KSTR halfs). Matches CDNA5 16-bit A layout:
//   lanes 0-15 : row = lane,    K = {0..7, 16..23}
//   lanes 16-31: row = lane-16, K = {8..15, 24..31}
// B tiles are stored N-major ([N][K]) in LDS so the same loader serves B.
// ---------------------------------------------------------------------------
__device__ __forceinline__ v16bf load_frag(const __bf16* tile, int lane) {
    const __bf16* p = tile + (lane & 15) * KSTR + ((lane >> 4) << 3);
    v8bf lo = *(const v8bf*)(p);
    v8bf hi = *(const v8bf*)(p + 16);
    v16bf r;
#pragma unroll
    for (int i = 0; i < 8; ++i) { r[i] = lo[i]; r[8 + i] = hi[i]; }
    return r;
}

// ---------------------------------------------------------------------------
// 1) zero per-expert counters
// ---------------------------------------------------------------------------
__global__ void init_counts_kernel(int* __restrict__ counts) {
    if (threadIdx.x < Eq) counts[threadIdx.x] = 0;
}

// ---------------------------------------------------------------------------
// 2) fp32 -> bf16 activation copy (one-time, halves GEMM1 A traffic)
// ---------------------------------------------------------------------------
__global__ void xcvt_kernel(const float* __restrict__ x, __bf16* __restrict__ xbf) {
    int i = blockIdx.x * blockDim.x + threadIdx.x;
    xbf[i] = (__bf16)x[i];
}

// ---------------------------------------------------------------------------
// 3) router: logits = x @ gate_w, top-2 (first-index tie-break), softmax of the
//    two selected logits, atomic compaction into per-expert slot lists.
//    slotcode = token*2 + k so downstream writes are slot-indexed (deterministic).
// ---------------------------------------------------------------------------
__global__ __launch_bounds__(256) void router_kernel(
    const float* __restrict__ x, const float* __restrict__ gw,
    int* __restrict__ counts, int* __restrict__ lists, float* __restrict__ wlist) {
    __shared__ float g[Dq * Eq];  // 32 KB
    for (int i = threadIdx.x; i < Dq * Eq; i += 256) g[i] = gw[i];
    __syncthreads();

    const int t = blockIdx.x * 256 + threadIdx.x;  // token id
    const float* xr = x + (size_t)t * Dq;

    float l[Eq];
#pragma unroll
    for (int e = 0; e < Eq; ++e) l[e] = 0.f;
    for (int d = 0; d < Dq; ++d) {
        const float xv = xr[d];
#pragma unroll
        for (int e = 0; e < Eq; ++e) l[e] += xv * g[d * Eq + e];
    }

    int e0 = 0; float l0 = l[0];
#pragma unroll
    for (int e = 1; e < Eq; ++e) if (l[e] > l0) { l0 = l[e]; e0 = e; }
    int e1 = -1; float l1 = -INFINITY;
#pragma unroll
    for (int e = 0; e < Eq; ++e) if (e != e0 && l[e] > l1) { l1 = l[e]; e1 = e; }

    const float z  = __expf(l1 - l0);     // <= 1, safe
    const float s  = 1.f + z;
    const float w0 = 1.f / s;
    const float w1 = z / s;

    int p0 = atomicAdd(&counts[e0], 1);
    lists[e0 * Tq + p0] = t * 2 + 0;
    wlist[e0 * Tq + p0] = w0;
    int p1 = atomicAdd(&counts[e1], 1);
    lists[e1 * Tq + p1] = t * 2 + 1;
    wlist[e1 * Tq + p1] = w1;
}

// ---------------------------------------------------------------------------
// Stage one gathered A row (32 bf16 = 64B) into LDS: async global->LDS when
// available (no VGPR round-trip, tracked by ASYNCcnt), else VGPR staging.
// Invalid rows are zero-filled through DS stores.
// ---------------------------------------------------------------------------
__device__ __forceinline__ void stage_a_row(__bf16* dst, const __bf16* src,
                                            const v8bf& z8) {
    if (src) {
#ifdef HAVE_ASYNC_LDS
#pragma unroll
        for (int c = 0; c < 4; ++c)
            __builtin_amdgcn_global_load_async_to_lds_b128(GPTR(src + c * 8),
                                                           LPTR(dst + c * 8), 0, 0);
#else
#pragma unroll
        for (int c = 0; c < 4; ++c) *(v8bf*)(dst + c * 8) = *(const v8bf*)(src + c * 8);
#endif
    } else {
#pragma unroll
        for (int c = 0; c < 4; ++c) *(v8bf*)(dst + c * 8) = z8;
    }
}

__device__ __forceinline__ void stage_fence() {
#ifdef HAVE_ASYNC_LDS
    wait_async0();
#endif
    __syncthreads();
}

// ---------------------------------------------------------------------------
// 4) grouped GEMM1 + gating: for expert e, tokens gathered via list,
//    h[slot][f] = silu(x@wi0) * (x@wi1) * route_w   (bf16 out)
//    Dual-WMMA: wi0 and wi1 share the A tile.
// ---------------------------------------------------------------------------
__global__ __launch_bounds__(128) void moe_gemm1_kernel(
    const __bf16* __restrict__ xbf,
    const float* __restrict__ wi0, const float* __restrict__ wi1,
    const int* __restrict__ counts, const int* __restrict__ lists,
    const float* __restrict__ wlist, __bf16* __restrict__ hbuf) {
    __shared__ __bf16 ldsA[BLK_M * KSTR];
    __shared__ __bf16 ldsB0[BLK_N * KSTR];
    __shared__ __bf16 ldsB1[BLK_N * KSTR];
    __shared__ int   sSlot[BLK_M];
    __shared__ float sW[BLK_M];

    const int NT = Fq / BLK_N;            // 64 n-tiles
    const int MT = Tq / BLK_M;            // 64 m-tiles (capacity)
    const int bid = blockIdx.x;
    const int e  = bid / (MT * NT);
    const int r  = bid - e * (MT * NT);
    const int mt = r / NT;
    const int nt = r - mt * NT;
    const int cnt = counts[e];
    const int m0  = mt * BLK_M;
    if (m0 >= cnt) return;                // uniform early exit

    const int tid = threadIdx.x;
    if (tid < BLK_M) {
        const int row = m0 + tid;
        if (row < cnt) { sSlot[tid] = lists[e * Tq + row]; sW[tid] = wlist[e * Tq + row]; }
        else           { sSlot[tid] = -1;                  sW[tid] = 0.f; }
    }
    __syncthreads();

    const int slotcode = sSlot[tid];
    const __bf16* arow = (slotcode >= 0) ? (xbf + (size_t)(slotcode >> 1) * Dq) : nullptr;

    const int n0 = nt * BLK_N;
    const float* w0p = wi0 + (size_t)e * Dq * Fq;
    const float* w1p = wi1 + (size_t)e * Dq * Fq;

    const int lane = tid & 31;
    const int wid  = tid >> 5;
    const int wm   = (wid >> 1) * 64;     // wave M offset (0/64)
    const int wn   = (wid & 1)  * 32;     // wave N offset (0/32)

    const v8f vzero = {0.f, 0.f, 0.f, 0.f, 0.f, 0.f, 0.f, 0.f};
    v8f acc0[4][2], acc1[4][2];
#pragma unroll
    for (int i = 0; i < 4; ++i)
#pragma unroll
        for (int j = 0; j < 2; ++j) { acc0[i][j] = vzero; acc1[i][j] = vzero; }

    v8bf z8;
#pragma unroll
    for (int i = 0; i < 8; ++i) z8[i] = (__bf16)0.f;

    const int tf = (tid & 15) * 4;        // f-offset within n-tile (0..60)
    const int td = (tid >> 4) * 4;        // d-offset within k-tile (0..28)

    for (int k0 = 0; k0 < Dq; k0 += BLK_K) {
        // ---- stage A (row-gathered bf16, async -> LDS when available) ----
        stage_a_row(&ldsA[tid * KSTR], arow ? arow + k0 : nullptr, z8);

        // ---- stage B0/B1: fp32 global -> bf16 LDS, transposed to [N][K] ----
        v4f q0[4], q1[4];
#pragma unroll
        for (int dd = 0; dd < 4; ++dd) {
            q0[dd] = *(const v4f*)(w0p + (size_t)(k0 + td + dd) * Fq + n0 + tf);
            q1[dd] = *(const v4f*)(w1p + (size_t)(k0 + td + dd) * Fq + n0 + tf);
        }
#pragma unroll
        for (int c = 0; c < 4; ++c) {     // f = tf + c ; pack 4 d-values -> b64 store
            v4bf p0, p1;
#pragma unroll
            for (int dd = 0; dd < 4; ++dd) { p0[dd] = (__bf16)q0[dd][c]; p1[dd] = (__bf16)q1[dd][c]; }
            *(v4bf*)&ldsB0[(tf + c) * KSTR + td] = p0;
            *(v4bf*)&ldsB1[(tf + c) * KSTR + td] = p1;
        }
        // prefetch next k-step's weight tile rows into cache
        if (k0 + BLK_K < Dq) {
            __builtin_prefetch(w0p + (size_t)(k0 + BLK_K + td) * Fq + n0 + tf, 0, 3);
            __builtin_prefetch(w1p + (size_t)(k0 + BLK_K + td) * Fq + n0 + tf, 0, 3);
        }
        stage_fence();

        v16bf af[4];
#pragma unroll
        for (int i = 0; i < 4; ++i) af[i] = load_frag(&ldsA[(wm + i * 16) * KSTR], lane);
#pragma unroll
        for (int j = 0; j < 2; ++j) {
            const v16bf b0 = load_frag(&ldsB0[(wn + j * 16) * KSTR], lane);
            const v16bf b1 = load_frag(&ldsB1[(wn + j * 16) * KSTR], lane);
#pragma unroll
            for (int i = 0; i < 4; ++i) {
                acc0[i][j] = __builtin_amdgcn_wmma_f32_16x16x32_bf16(
                    false, af[i], false, b0, (short)0, acc0[i][j], false, false);
                acc1[i][j] = __builtin_amdgcn_wmma_f32_16x16x32_bf16(
                    false, af[i], false, b1, (short)0, acc1[i][j], false, false);
            }
        }
        __syncthreads();
    }

    // ---- epilogue: h = silu(acc0) * acc1 * route_w, bf16 store (slot rows) ----
    const int nlane = lane & 15;
    const int mhalf = (lane >> 4) * 8;
#pragma unroll
    for (int i = 0; i < 4; ++i) {
#pragma unroll
        for (int j = 0; j < 2; ++j) {
            const int fcol = n0 + wn + j * 16 + nlane;
#pragma unroll
            for (int v = 0; v < 8; ++v) {
                const int m    = wm + i * 16 + mhalf + v;
                const int slot = sSlot[m];
                if (slot >= 0) {
                    const float a = acc0[i][j][v];
                    const float g = a / (1.f + __expf(-a));
                    const float h = g * acc1[i][j][v] * sW[m];
                    hbuf[(size_t)slot * Fq + fcol] = (__bf16)h;
                }
            }
        }
    }
}

// ---------------------------------------------------------------------------
// 5) grouped GEMM2: out_slots[slot][d] = h[slot] @ wo[e]   (fp32 out, no atomics)
// ---------------------------------------------------------------------------
__global__ __launch_bounds__(128) void moe_gemm2_kernel(
    const __bf16* __restrict__ hbuf, const float* __restrict__ wo,
    const int* __restrict__ counts, const int* __restrict__ lists,
    float* __restrict__ oslots) {
    __shared__ __bf16 ldsA[BLK_M * KSTR];
    __shared__ __bf16 ldsB[BLK_N * KSTR];
    __shared__ int sSlot[BLK_M];

    const int NT = Dq / BLK_N;            // 16
    const int MT = Tq / BLK_M;            // 64
    const int bid = blockIdx.x;
    const int e  = bid / (MT * NT);
    const int r  = bid - e * (MT * NT);
    const int mt = r / NT;
    const int nt = r - mt * NT;
    const int cnt = counts[e];
    const int m0  = mt * BLK_M;
    if (m0 >= cnt) return;

    const int tid = threadIdx.x;
    if (tid < BLK_M) {
        const int row = m0 + tid;
        sSlot[tid] = (row < cnt) ? lists[e * Tq + row] : -1;
    }
    __syncthreads();

    const int slotcode = sSlot[tid];
    const __bf16* arow = (slotcode >= 0) ? (hbuf + (size_t)slotcode * Fq) : nullptr;

    const int n0 = nt * BLK_N;
    const float* wp = wo + (size_t)e * Fq * Dq;

    const int lane = tid & 31;
    const int wid  = tid >> 5;
    const int wm   = (wid >> 1) * 64;
    const int wn   = (wid & 1)  * 32;

    const v8f vzero = {0.f, 0.f, 0.f, 0.f, 0.f, 0.f, 0.f, 0.f};
    v8f acc[4][2];
#pragma unroll
    for (int i = 0; i < 4; ++i)
#pragma unroll
        for (int j = 0; j < 2; ++j) acc[i][j] = vzero;

    v8bf z8;
#pragma unroll
    for (int i = 0; i < 8; ++i) z8[i] = (__bf16)0.f;

    const int tf = (tid & 15) * 4;
    const int td = (tid >> 4) * 4;

    for (int k0 = 0; k0 < Fq; k0 += BLK_K) {
        stage_a_row(&ldsA[tid * KSTR], arow ? arow + k0 : nullptr, z8);

        v4f q0[4];
#pragma unroll
        for (int dd = 0; dd < 4; ++dd)
            q0[dd] = *(const v4f*)(wp + (size_t)(k0 + td + dd) * Dq + n0 + tf);
#pragma unroll
        for (int c = 0; c < 4; ++c) {
            v4bf p0;
#pragma unroll
            for (int dd = 0; dd < 4; ++dd) p0[dd] = (__bf16)q0[dd][c];
            *(v4bf*)&ldsB[(tf + c) * KSTR + td] = p0;
        }
        if (k0 + BLK_K < Fq) {
            __builtin_prefetch(wp + (size_t)(k0 + BLK_K + td) * Dq + n0 + tf, 0, 3);
        }
        stage_fence();

        v16bf af[4];
#pragma unroll
        for (int i = 0; i < 4; ++i) af[i] = load_frag(&ldsA[(wm + i * 16) * KSTR], lane);
#pragma unroll
        for (int j = 0; j < 2; ++j) {
            const v16bf bf = load_frag(&ldsB[(wn + j * 16) * KSTR], lane);
#pragma unroll
            for (int i = 0; i < 4; ++i) {
                acc[i][j] = __builtin_amdgcn_wmma_f32_16x16x32_bf16(
                    false, af[i], false, bf, (short)0, acc[i][j], false, false);
            }
        }
        __syncthreads();
    }

    const int nlane = lane & 15;
    const int mhalf = (lane >> 4) * 8;
#pragma unroll
    for (int i = 0; i < 4; ++i) {
#pragma unroll
        for (int j = 0; j < 2; ++j) {
            const int col = n0 + wn + j * 16 + nlane;
#pragma unroll
            for (int v = 0; v < 8; ++v) {
                const int m    = wm + i * 16 + mhalf + v;
                const int slot = sSlot[m];
                if (slot >= 0) oslots[(size_t)slot * Dq + col] = acc[i][j][v];
            }
        }
    }
}

// ---------------------------------------------------------------------------
// 6) reduce the K=2 slot contributions per token
// ---------------------------------------------------------------------------
__global__ void reduce_kernel(const float* __restrict__ oslots, float* __restrict__ out) {
    const int i = blockIdx.x * blockDim.x + threadIdx.x;  // token*Dq + d
    const int t = i >> 10;                                 // Dq == 1024
    const int d = i & 1023;
    const size_t base = (size_t)(t * 2) * Dq + d;
    out[i] = oslots[base] + oslots[base + Dq];
}

// ---------------------------------------------------------------------------
extern "C" void kernel_launch(void* const* d_in, const int* in_sizes, int n_in,
                              void* d_out, int out_size, void* d_ws, size_t ws_size,
                              hipStream_t stream) {
    (void)in_sizes; (void)n_in; (void)out_size; (void)ws_size;
    const float* x   = (const float*)d_in[0];   // [B,S,D]
    const float* gw  = (const float*)d_in[1];   // [D,E]
    const float* wi0 = (const float*)d_in[2];   // [E,D,F]
    const float* wi1 = (const float*)d_in[3];   // [E,D,F]
    const float* wo  = (const float*)d_in[4];   // [E,F,D]
    float* out = (float*)d_out;                 // [B,S,D]

    // workspace carve-up
    char* ws = (char*)d_ws;
    size_t off = 0;
    int* counts = (int*)(ws + off);              off += 256;                       // E ints (padded)
    int* lists  = (int*)(ws + off);              off += (size_t)Eq * Tq * 4;       // slot codes
    float* wlist = (float*)(ws + off);           off += (size_t)Eq * Tq * 4;       // route weights
    __bf16* xbf  = (__bf16*)(ws + off);          off += (size_t)Tq * Dq * 2;       // bf16 activations
    __bf16* hbuf = (__bf16*)(ws + off);          off += (size_t)NSLOT * Fq * 2;    // gated hidden
    float* oslots = (float*)(ws + off);          off += (size_t)NSLOT * Dq * 4;    // per-slot output

    init_counts_kernel<<<1, 32, 0, stream>>>(counts);
    xcvt_kernel<<<(Tq * Dq) / 256, 256, 0, stream>>>(x, xbf);
    router_kernel<<<Tq / 256, 256, 0, stream>>>(x, gw, counts, lists, wlist);
    moe_gemm1_kernel<<<Eq * (Tq / BLK_M) * (Fq / BLK_N), 128, 0, stream>>>(
        xbf, wi0, wi1, counts, lists, wlist, hbuf);
    moe_gemm2_kernel<<<Eq * (Tq / BLK_M) * (Dq / BLK_N), 128, 0, stream>>>(
        hbuf, wo, counts, lists, oslots);
    reduce_kernel<<<(Tq * Dq) / 256, 256, 0, stream>>>(oslots, out);
}
